// DGCN_41875931136705
// MI455X (gfx1250) — matrix operand order repaired and checked by reading, not compiled, not run on hardware
//
#include <hip/hip_runtime.h>

// Problem constants (match reference)
#define BB 4
#define NN 16384            // 2^14
#define KK 8
#define EE (NN*KK)          // 131072 = 2^17
#define BNROWS (BB*NN)      // 65536
#define BEROWS (BB*EE)      // 524288

typedef __attribute__((ext_vector_type(16))) _Float16 v16h;
typedef __attribute__((ext_vector_type(8)))  float    v8f;

__global__ void zero_f32(float* p, int n) {
    int i = blockIdx.x * blockDim.x + threadIdx.x;
    if (i < n) p[i] = 0.f;
}

// ---------------------------------------------------------------------------
// Y = A @ W via v_wmma_f32_16x16x32_f16 with fused per-column BN statistics.
// Compile-time CIN/COUT -> branch-free b128 A loads, fully unrolled K loop.
// Weight tile staged to LDS once per block (shared by all 4 waves), B fragments
// read back as two ds_load_b128 per lane.
// MODE==0: A is [R x CIN] row-major f32.
// MODE==1: row r is x[nbr(r)] - x[ctr(r)] (CIN==8) built from X + edge_index.
// ---------------------------------------------------------------------------
template<int CIN, int COUT, int MODE>
__global__ void __launch_bounds__(128)
gemm_wmma(const float* __restrict__ A, const float* __restrict__ W,
          const int* __restrict__ ei, const float* __restrict__ X,
          float* __restrict__ Y, float* __restrict__ stats)
{
    constexpr int KP = (CIN + 31) & ~31;        // K padded to WMMA step
    __shared__ _Float16 wl[16 * KP];            // [n][k], rows 64B aligned

    const int tid = threadIdx.x;
    const int nt  = blockIdx.y;

    // Cooperative W-tile staging (f32 -> f16), zero-padded in K.
    #pragma unroll
    for (int idx = 0; idx < 16 * KP / 128; ++idx) {
        const int j = idx * 128 + tid;
        const int n = j / KP;
        const int k = j & (KP - 1);
        const float wv = (k < CIN) ? W[(size_t)k * COUT + (nt << 4) + n] : 0.f;
        wl[j] = (_Float16)wv;
    }
    __syncthreads();

    const int lane = tid & 31;
    const int wave = tid >> 5;
    const int row0 = (blockIdx.x * 4 + wave) << 4;   // grids are exact multiples
    const int half = lane >> 4;                      // 0 or 1
    const int lm   = lane & 15;
    const int m    = row0 + lm;                      // A row held by this lane
    const int n    = (nt << 4) + lm;                 // output column of this lane

    const float* arow;
    const float* nrow = nullptr;
    if (MODE == 0) {
        arow = A + (size_t)m * CIN;
    } else {
        const int bb = m >> 17;                      // / EE
        const int e  = m & (EE - 1);
        const int* eb = ei + (size_t)bb * 2 * EE;
        arow = X + ((size_t)bb * NN + eb[e]) * 8;        // center row
        nrow = X + ((size_t)bb * NN + eb[EE + e]) * 8;   // neighbor row
    }

    v8f acc = {};
    #pragma unroll
    for (int k0 = 0; k0 < KP; k0 += 32) {
        v16h a;
        if (CIN == 8) {
            // Only K<8 contributes; lanes with half==1 hold K>=8 -> all zero.
            #pragma unroll
            for (int e2 = 0; e2 < 16; ++e2) a[e2] = (_Float16)0.f;
            if (half == 0) {
                const float4 c0 = *(const float4*)(arow);
                const float4 c1 = *(const float4*)(arow + 4);
                if (MODE == 1) {
                    const float4 d0 = *(const float4*)(nrow);
                    const float4 d1 = *(const float4*)(nrow + 4);
                    a[0] = (_Float16)(d0.x - c0.x); a[1] = (_Float16)(d0.y - c0.y);
                    a[2] = (_Float16)(d0.z - c0.z); a[3] = (_Float16)(d0.w - c0.w);
                    a[4] = (_Float16)(d1.x - c1.x); a[5] = (_Float16)(d1.y - c1.y);
                    a[6] = (_Float16)(d1.z - c1.z); a[7] = (_Float16)(d1.w - c1.w);
                } else {
                    a[0] = (_Float16)c0.x; a[1] = (_Float16)c0.y;
                    a[2] = (_Float16)c0.z; a[3] = (_Float16)c0.w;
                    a[4] = (_Float16)c1.x; a[5] = (_Float16)c1.y;
                    a[6] = (_Float16)c1.z; a[7] = (_Float16)c1.w;
                }
            }
        } else {
            // One contiguous 16-float chunk: K in [k0+16h, +16).
            const float4 c0 = *(const float4*)(arow + k0 + half * 16);
            const float4 c1 = *(const float4*)(arow + k0 + half * 16 + 4);
            const float4 c2 = *(const float4*)(arow + k0 + half * 16 + 8);
            const float4 c3 = *(const float4*)(arow + k0 + half * 16 + 12);
            a[0]  = (_Float16)c0.x; a[1]  = (_Float16)c0.y;
            a[2]  = (_Float16)c0.z; a[3]  = (_Float16)c0.w;
            a[4]  = (_Float16)c1.x; a[5]  = (_Float16)c1.y;
            a[6]  = (_Float16)c1.z; a[7]  = (_Float16)c1.w;
            a[8]  = (_Float16)c2.x; a[9]  = (_Float16)c2.y;
            a[10] = (_Float16)c2.z; a[11] = (_Float16)c2.w;
            a[12] = (_Float16)c3.x; a[13] = (_Float16)c3.y;
            a[14] = (_Float16)c3.z; a[15] = (_Float16)c3.w;
        }
        // B fragment: column lm, K = k0 + 16*half + e -> 16 contiguous halfs (32B aligned)
        const v16h w = *(const v16h*)&wl[lm * KP + k0 + half * 16];
        acc = __builtin_amdgcn_wmma_f32_16x16x32_f16(false, a, false, w,
                                                     (short)0, acc, false, false);
    }

    // Epilogue: store f32 tile + fused per-column BN statistics.
    float s = 0.f, sq = 0.f;
    #pragma unroll
    for (int r = 0; r < 8; ++r) {
        const int row = row0 + half * 8 + r;
        const float v = acc[r];
        Y[(size_t)row * COUT + n] = v;
        s  += v;
        sq += v * v;
    }
    s  += __shfl_xor(s, 16);
    sq += __shfl_xor(sq, 16);
    if (half == 0) {
        atomicAdd(&stats[n], s);
        atomicAdd(&stats[COUT + n], sq);
    }
}

// scale = g * rsqrt(var + eps), shift = b - scale * mean
__global__ void bn_finalize(const float* __restrict__ stats,
                            const float* __restrict__ g, const float* __restrict__ b,
                            float* __restrict__ ss, int C, float invR)
{
    int c = blockIdx.x * blockDim.x + threadIdx.x;
    if (c >= C) return;
    const float m = stats[c] * invR;
    const float v = stats[C + c] * invR - m * m;
    const float a = g[c] * rsqrtf(v + 1e-5f);
    ss[c]     = a;
    ss[C + c] = b[c] - a * m;
}

__device__ __forceinline__ float lrelu(float v) { return v > 0.f ? v : 0.2f * v; }

// y = leaky_relu(scale*y + shift) in place, 4 elements/thread; C = 1<<cshift >= 4
__global__ void bn_apply4(float* __restrict__ y, const float* __restrict__ ss,
                          int cshift, int total4)
{
    int i = blockIdx.x * blockDim.x + threadIdx.x;
    if (i >= total4) return;
    const int C = 1 << cshift;
    const int c = (i << 2) & (C - 1);            // aligned, no wrap within 4
    float4 v = ((const float4*)y)[i];
    v.x = lrelu(fmaf(ss[c + 0], v.x, ss[C + c + 0]));
    v.y = lrelu(fmaf(ss[c + 1], v.y, ss[C + c + 1]));
    v.z = lrelu(fmaf(ss[c + 2], v.z, ss[C + c + 2]));
    v.w = lrelu(fmaf(ss[c + 3], v.w, ss[C + c + 3]));
    ((float4*)y)[i] = v;
}

__device__ __forceinline__ void max4(float4& m, const float4 v) {
    m.x = fmaxf(m.x, v.x); m.y = fmaxf(m.y, v.y);
    m.z = fmaxf(m.z, v.z); m.w = fmaxf(m.w, v.w);
}

// xa[i,c:c+4] += max_k yb[nbr(i,k), c:c+4]   (layers 1..3; yb has node rows)
__global__ void segmax_add4(float* __restrict__ xa, const float* __restrict__ yb,
                            const int* __restrict__ ei, int cshift)
{
    int idx = blockIdx.x * blockDim.x + threadIdx.x;   // BNROWS * C/4 threads
    const int C  = 1 << cshift;
    const int i  = idx >> (cshift - 2);
    if (i >= BNROWS) return;
    const int c  = (idx << 2) & (C - 1);
    const int b  = i >> 14;                 // / NN
    const int nd = i & (NN - 1);
    const int* nb = ei + (size_t)b * 2 * EE + EE + (size_t)nd * KK;
    float4 mx = make_float4(-3.4e38f, -3.4e38f, -3.4e38f, -3.4e38f);
    #pragma unroll
    for (int k = 0; k < KK; ++k) {
        const int j = nb[k];
        max4(mx, *(const float4*)&yb[((size_t)(b * NN + j) << cshift) + c]);
    }
    float4* dst = (float4*)&xa[((size_t)i << cshift) + c];
    float4 d = *dst;
    d.x += mx.x; d.y += mx.y; d.z += mx.z; d.w += mx.w;
    *dst = d;
}

// Layer 0: yb indexed by edge rows; centers are structurally n = e/K.
__global__ void segmax0_add4(float* __restrict__ xa, const float* __restrict__ yb,
                             int cshift)
{
    int idx = blockIdx.x * blockDim.x + threadIdx.x;
    const int C  = 1 << cshift;
    const int i  = idx >> (cshift - 2);
    if (i >= BNROWS) return;
    const int c  = (idx << 2) & (C - 1);
    const int b  = i >> 14;
    const int nd = i & (NN - 1);
    const size_t rbase = (size_t)b * EE + (size_t)nd * KK;
    float4 mx = make_float4(-3.4e38f, -3.4e38f, -3.4e38f, -3.4e38f);
    #pragma unroll
    for (int k = 0; k < KK; ++k)
        max4(mx, *(const float4*)&yb[((rbase + k) << cshift) + c]);
    float4* dst = (float4*)&xa[((size_t)i << cshift) + c];
    float4 d = *dst;
    d.x += mx.x; d.y += mx.y; d.z += mx.z; d.w += mx.w;
    *dst = d;
}

__device__ __forceinline__ float dot4n(const float4 a, const float4 b, float s) {
    s = fmaf(a.x, b.x, s); s = fmaf(a.y, b.y, s);
    s = fmaf(a.z, b.z, s); return fmaf(a.w, b.w, s);
}

// y_cat = concat(x0,x1,x2,x3) @ Wcat (480 -> 1) with LDS block-reduced stats
__global__ void cat_dot(const float* __restrict__ x0, const float* __restrict__ x1,
                        const float* __restrict__ x2, const float* __restrict__ x3,
                        const float* __restrict__ Wc, float* __restrict__ yc,
                        float* __restrict__ stats)
{
    const int i = blockIdx.x * blockDim.x + threadIdx.x;   // < BNROWS
    float s = 0.f;
    const float4* w4 = (const float4*)Wc;
    const float4* p4 = (const float4*)(x0 + (size_t)i * 32);
    #pragma unroll
    for (int c = 0; c < 8; ++c)  s = dot4n(p4[c], w4[c], s);
    p4 = (const float4*)(x1 + (size_t)i * 64);
    #pragma unroll
    for (int c = 0; c < 16; ++c) s = dot4n(p4[c], w4[8 + c], s);
    p4 = (const float4*)(x2 + (size_t)i * 128);
    #pragma unroll
    for (int c = 0; c < 32; ++c) s = dot4n(p4[c], w4[24 + c], s);
    p4 = (const float4*)(x3 + (size_t)i * 256);
    #pragma unroll
    for (int c = 0; c < 64; ++c) s = dot4n(p4[c], w4[56 + c], s);
    yc[i] = s;

    __shared__ float shs[256];
    __shared__ float shq[256];
    const int t = threadIdx.x;
    shs[t] = s; shq[t] = s * s;
    __syncthreads();
    for (int off = 128; off > 0; off >>= 1) {
        if (t < off) { shs[t] += shs[t + off]; shq[t] += shq[t + off]; }
        __syncthreads();
    }
    if (t == 0) { atomicAdd(&stats[0], shs[0]); atomicAdd(&stats[1], shq[0]); }
}

__global__ void cat_out(const float* __restrict__ yc, const float* __restrict__ ss,
                        const float* __restrict__ bias, float* __restrict__ out)
{
    int i = blockIdx.x * blockDim.x + threadIdx.x;
    if (i >= BNROWS) return;
    out[i] = lrelu(fmaf(ss[0], yc[i], ss[1])) + bias[0];
}

extern "C" void kernel_launch(void* const* d_in, const int* in_sizes, int n_in,
                              void* d_out, int out_size, void* d_ws, size_t ws_size,
                              hipStream_t stream)
{
    (void)in_sizes; (void)n_in; (void)out_size; (void)ws_size;
    const float* x   = (const float*)d_in[0];
    const int*   ei  = (const int*)  d_in[1];
    const float* W0a = (const float*)d_in[2],  *g0a = (const float*)d_in[3],  *b0a = (const float*)d_in[4];
    const float* W0b = (const float*)d_in[5],  *g0b = (const float*)d_in[6],  *b0b = (const float*)d_in[7];
    const float* W1a = (const float*)d_in[8],  *g1a = (const float*)d_in[9],  *b1a = (const float*)d_in[10];
    const float* W1b = (const float*)d_in[11], *g1b = (const float*)d_in[12], *b1b = (const float*)d_in[13];
    const float* W2a = (const float*)d_in[14], *g2a = (const float*)d_in[15], *b2a = (const float*)d_in[16];
    const float* W2b = (const float*)d_in[17], *g2b = (const float*)d_in[18], *b2b = (const float*)d_in[19];
    const float* W3a = (const float*)d_in[20], *g3a = (const float*)d_in[21], *b3a = (const float*)d_in[22];
    const float* W3b = (const float*)d_in[23], *g3b = (const float*)d_in[24], *b3b = (const float*)d_in[25];
    const float* Wct = (const float*)d_in[26], *gct = (const float*)d_in[27], *bct = (const float*)d_in[28];
    const float* bias= (const float*)d_in[29];

    // Workspace layout (floats). BEROWS*32 == BNROWS*256, so one shared b-branch
    // scratch buffer covers every layer. ~193 MB total.
    float* ws  = (float*)d_ws;
    float* x0  = ws;                               // BNROWS*32   (y0a in place)
    float* x1  = x0 + (size_t)BNROWS * 32;         // BNROWS*64
    float* x2  = x1 + (size_t)BNROWS * 64;         // BNROWS*128
    float* x3  = x2 + (size_t)BNROWS * 128;        // BNROWS*256
    float* yb  = x3 + (size_t)BNROWS * 256;        // max(BEROWS*32, BNROWS*256)
    float* yc  = yb + (size_t)BNROWS * 256;        // BNROWS
    float* stA = yc + BNROWS;                      // 512
    float* stB = stA + 512;                        // 512
    float* ssA = stB + 512;                        // 512
    float* ssB = ssA + 512;                        // 512

    const float invN = 1.f / (float)BNROWS;
    const float invE = 1.f / (float)BEROWS;

    auto zero = [&](float* p, int n) {
        zero_f32<<<(n + 127) / 128, 128, 0, stream>>>(p, n);
    };

    // ---------------- layer 0: 8 -> 32 ----------------
    zero(stA, 64); zero(stB, 64);
    gemm_wmma<8, 32, 0><<<dim3(BNROWS / 64, 2), 128, 0, stream>>>(x, W0a, ei, x, x0, stA);
    gemm_wmma<8, 32, 1><<<dim3(BEROWS / 64, 2), 128, 0, stream>>>(nullptr, W0b, ei, x, yb, stB);
    bn_finalize<<<1, 64, 0, stream>>>(stA, g0a, b0a, ssA, 32, invN);
    bn_finalize<<<1, 64, 0, stream>>>(stB, g0b, b0b, ssB, 32, invE);
    bn_apply4<<<(BNROWS * 32 / 4) / 256, 256, 0, stream>>>(x0, ssA, 5, BNROWS * 32 / 4);
    bn_apply4<<<(BEROWS * 32 / 4) / 256, 256, 0, stream>>>(yb, ssB, 5, BEROWS * 32 / 4);
    segmax0_add4<<<(BNROWS * 32 / 4) / 256, 256, 0, stream>>>(x0, yb, 5);

    // ---------------- layer 1: 32 -> 64 ----------------
    zero(stA, 128); zero(stB, 128);
    gemm_wmma<32, 64, 0><<<dim3(BNROWS / 64, 4), 128, 0, stream>>>(x0, W1a, ei, x, x1, stA);
    gemm_wmma<32, 64, 0><<<dim3(BNROWS / 64, 4), 128, 0, stream>>>(x0, W1b, ei, x, yb, stB);
    bn_finalize<<<1, 64, 0, stream>>>(stA, g1a, b1a, ssA, 64, invN);
    bn_finalize<<<1, 64, 0, stream>>>(stB, g1b, b1b, ssB, 64, invN);
    bn_apply4<<<(BNROWS * 64 / 4) / 256, 256, 0, stream>>>(x1, ssA, 6, BNROWS * 64 / 4);
    bn_apply4<<<(BNROWS * 64 / 4) / 256, 256, 0, stream>>>(yb, ssB, 6, BNROWS * 64 / 4);
    segmax_add4<<<(BNROWS * 64 / 4) / 256, 256, 0, stream>>>(x1, yb, ei, 6);

    // ---------------- layer 2: 64 -> 128 ----------------
    zero(stA, 256); zero(stB, 256);
    gemm_wmma<64, 128, 0><<<dim3(BNROWS / 64, 8), 128, 0, stream>>>(x1, W2a, ei, x, x2, stA);
    gemm_wmma<64, 128, 0><<<dim3(BNROWS / 64, 8), 128, 0, stream>>>(x1, W2b, ei, x, yb, stB);
    bn_finalize<<<1, 128, 0, stream>>>(stA, g2a, b2a, ssA, 128, invN);
    bn_finalize<<<1, 128, 0, stream>>>(stB, g2b, b2b, ssB, 128, invN);
    bn_apply4<<<(BNROWS * 128 / 4) / 256, 256, 0, stream>>>(x2, ssA, 7, BNROWS * 128 / 4);
    bn_apply4<<<(BNROWS * 128 / 4) / 256, 256, 0, stream>>>(yb, ssB, 7, BNROWS * 128 / 4);
    segmax_add4<<<(BNROWS * 128 / 4) / 256, 256, 0, stream>>>(x2, yb, ei, 7);

    // ---------------- layer 3: 128 -> 256 ----------------
    zero(stA, 512); zero(stB, 512);
    gemm_wmma<128, 256, 0><<<dim3(BNROWS / 64, 16), 128, 0, stream>>>(x2, W3a, ei, x, x3, stA);
    gemm_wmma<128, 256, 0><<<dim3(BNROWS / 64, 16), 128, 0, stream>>>(x2, W3b, ei, x, yb, stB);
    bn_finalize<<<1, 256, 0, stream>>>(stA, g3a, b3a, ssA, 256, invN);
    bn_finalize<<<1, 256, 0, stream>>>(stB, g3b, b3b, ssB, 256, invN);
    bn_apply4<<<(BNROWS * 256 / 4) / 256, 256, 0, stream>>>(x3, ssA, 8, BNROWS * 256 / 4);
    bn_apply4<<<(BNROWS * 256 / 4) / 256, 256, 0, stream>>>(yb, ssB, 8, BNROWS * 256 / 4);
    segmax_add4<<<(BNROWS * 256 / 4) / 256, 256, 0, stream>>>(x3, yb, ei, 8);

    // ---------------- cat head: 480 -> 1 ----------------
    zero(stA, 2);
    cat_dot<<<BNROWS / 256, 256, 0, stream>>>(x0, x1, x2, x3, Wct, yc, stA);
    bn_finalize<<<1, 32, 0, stream>>>(stA, gct, bct, ssA, 1, invN);
    cat_out<<<BNROWS / 256, 256, 0, stream>>>(yc, ssA, bias, (float*)d_out);
}
